// EquivariantDeepSetsEncoder_86131274154482
// MI455X (gfx1250) — matrix operand order repaired and checked by reading, not compiled, not run on hardware
//
#include <hip/hip_runtime.h>

// MI455X / gfx1250 implementation of EquivariantDeepSetsEncoder.
// P = exp(-dist) kept unnormalized in fp16 (67MB -> L2-resident on 192MB L2),
// fp32 row-sums; the three aggregation GEMMs P@H run on
// v_wmma_f32_16x16x32_f16 with 2x2 register tiling (4 WMMA per 4 fragment
// loads) and software pipelining (next K-tile loads issued before current
// WMMAs). Feature matrices stored d-major (transposed) so every fragment load
// is a contiguous global_load_b128. Required workspace ~90MB.

typedef _Float16 f16;
typedef _Float16 __attribute__((ext_vector_type(8)))  v8h;
typedef _Float16 __attribute__((ext_vector_type(16))) v16h;
typedef float    __attribute__((ext_vector_type(8)))  v8f;

#define BATCH  8
#define NPTS   2048
#define DIMS   3
#define LATENT 128

// ---------------------------------------------------------------- centroid
__global__ void centroid_kernel(const float* __restrict__ x,
                                const float* __restrict__ mask,
                                float* __restrict__ cent) {
    int b = blockIdx.x, t = threadIdx.x;
    float sx = 0.f, sy = 0.f, sz = 0.f, sc = 0.f;
    for (int n = t; n < NPTS; n += blockDim.x) {
        float m = mask[b * NPTS + n];
        const float* xp = x + ((size_t)b * NPTS + n) * DIMS;
        sx += xp[0] * m; sy += xp[1] * m; sz += xp[2] * m; sc += m;
    }
    __shared__ float r0[256], r1[256], r2[256], r3[256];
    r0[t] = sx; r1[t] = sy; r2[t] = sz; r3[t] = sc;
    __syncthreads();
    for (int s = blockDim.x >> 1; s > 0; s >>= 1) {
        if (t < s) { r0[t] += r0[t+s]; r1[t] += r1[t+s]; r2[t] += r2[t+s]; r3[t] += r3[t+s]; }
        __syncthreads();
    }
    if (t == 0) {
        float c = fmaxf(r3[0], 1.0f);
        cent[b*4+0] = r0[0] / c;
        cent[b*4+1] = r1[0] / c;
        cent[b*4+2] = r2[0] / c;
        cent[b*4+3] = r3[0];            // raw count for the readout pool
    }
}

// ------------------------------------- center points, build padded HT0 (f16)
// xc4: (B,N,4) fp32 for fast distance pass. ht0: (B,16,N) f16, rows 3..15 = 0.
__global__ void center_kernel(const float* __restrict__ x,
                              const float* __restrict__ cent,
                              float* __restrict__ xc4,
                              f16* __restrict__ ht0) {
    int idx = blockIdx.x * blockDim.x + threadIdx.x;  // 0 .. B*N-1
    int b = idx / NPTS, n = idx % NPTS;
    const float* xp = x + (size_t)idx * DIMS;
    float vx = xp[0] - cent[b*4+0];
    float vy = xp[1] - cent[b*4+1];
    float vz = xp[2] - cent[b*4+2];
    float4 v; v.x = vx; v.y = vy; v.z = vz; v.w = 0.0f;
    ((float4*)xc4)[idx] = v;
    f16* h = ht0 + (size_t)b * 16 * NPTS + n;
    h[0 * (size_t)NPTS] = (f16)vx;
    h[1 * (size_t)NPTS] = (f16)vy;
    h[2 * (size_t)NPTS] = (f16)vz;
#pragma unroll
    for (int d = 3; d < 16; ++d) h[(size_t)d * NPTS] = (f16)0.0f;
}

// -------------------------- pairwise dist -> P = exp(-d2) (f16) + rowsum S
// softmax(-dist) row max is 0 (diagonal), so exp(-d2)/sum is exact softmax.
__global__ void distexp_kernel(const float* __restrict__ xc4,
                               f16* __restrict__ P,
                               float* __restrict__ S) {
    int i = blockIdx.x, b = blockIdx.y, t = threadIdx.x;
    const float4* X = (const float4*)xc4 + (size_t)b * NPTS;
    float4 xi = X[i];
    f16* Prow = P + ((size_t)b * NPTS + i) * NPTS;
    float acc = 0.f;
    for (int j = t; j < NPTS; j += blockDim.x) {
        float4 xj = X[j];
        float dx = xi.x - xj.x, dy = xi.y - xj.y, dz = xi.z - xj.z;
        float p = __expf(-(dx*dx + dy*dy + dz*dz));
        Prow[j] = (f16)p;
        acc += p;
    }
    __shared__ float red[256];
    red[t] = acc; __syncthreads();
    for (int s = blockDim.x >> 1; s > 0; s >>= 1) {
        if (t < s) red[t] += red[t+s];
        __syncthreads();
    }
    if (t == 0) S[b * NPTS + i] = red[0];
}

// --------------------------------------------------------- WMMA helpers
__device__ __forceinline__ v16h load_frag_a(const f16* p) {
    // A 16x32 f16 fragment: two contiguous b128 chunks (K +0..7 and +16..23
    // for lanes 0-15; +8..15 / +24..31 for lanes 16-31 via laneHi*8 base).
    v8h lo = *(const v8h*)(p);
    v8h hi = *(const v8h*)(p + 16);
    return __builtin_shufflevector(lo, hi, 0,1,2,3,4,5,6,7,8,9,10,11,12,13,14,15);
}
__device__ __forceinline__ v16h load_frag_b(const f16* p) {
    // B 32x16 f16 fragment from transposed (d-major) storage: two b128 chunks
    // (K +0..7 and +16..23 for lanes 0-15; +8..15 / +24..31 via laneHi*8 base).
    v8h lo = *(const v8h*)(p);
    v8h hi = *(const v8h*)(p + 16);
    return __builtin_shufflevector(lo, hi, 0,1,2,3,4,5,6,7,8,9,10,11,12,13,14,15);
}
__device__ __forceinline__ v8f wmma16(v16h a, v16h b, v8f c) {
    return __builtin_amdgcn_wmma_f32_16x16x32_f16(
        /*neg_a=*/false, a, /*neg_b=*/false, b,
        /*c_mod=*/(short)0, c, /*reuse_a=*/false, /*reuse_b=*/false);
}

// ----------------------------- Hagg = P @ H  via v_wmma_f32_16x16x32_f16
// P: (B,N,N) f16 row-major. HT: (B,DPAD,N) f16 (features transposed).
// Hagg: (B,N,DPAD) fp32. Wave computes 32(M) x 16*NT(N); 4 waves/block -> M=128.
template<int DPAD, int NT>
__global__ void agg_wmma_kernel(const f16* __restrict__ P,
                                const f16* __restrict__ HT,
                                float* __restrict__ Hagg) {
    int wave   = threadIdx.x >> 5;
    int lane   = threadIdx.x & 31;
    int laneLo = lane & 15;
    int laneHi = lane >> 4;
    int b  = blockIdx.z;
    int m0 = (blockIdx.x * 4 + wave) * 32;
    int n0 = blockIdx.y * (16 * NT);

    const f16* pA0 = P  + ((size_t)b * NPTS + m0 + laneLo) * NPTS + laneHi * 8;
    const f16* pA1 = pA0 + (size_t)16 * NPTS;
    const f16* pB0 = HT + ((size_t)b * DPAD + n0 + laneLo) * NPTS + laneHi * 8;
    const f16* pB1 = pB0 + (size_t)16 * NPTS;

    v8f c00 = {}, c10 = {}, c01 = {}, c11 = {};

    // prologue: fragments for k = 0
    v16h a0 = load_frag_a(pA0);
    v16h a1 = load_frag_a(pA1);
    v16h b0 = load_frag_b(pB0);
    v16h b1 = {};
    if constexpr (NT == 2) b1 = load_frag_b(pB1);

    // software-pipelined K loop: issue k+32 loads, then WMMA on k fragments
    for (int k = 32; k < NPTS; k += 32) {
        v16h na0 = load_frag_a(pA0 + k);
        v16h na1 = load_frag_a(pA1 + k);
        v16h nb0 = load_frag_b(pB0 + k);
        v16h nb1 = {};
        if constexpr (NT == 2) nb1 = load_frag_b(pB1 + k);
        __builtin_prefetch((const void*)(pA0 + k + 512), 0, 3);  // speculative, branch-free

        c00 = wmma16(a0, b0, c00);
        c10 = wmma16(a1, b0, c10);
        if constexpr (NT == 2) {
            c01 = wmma16(a0, b1, c01);
            c11 = wmma16(a1, b1, c11);
        }
        a0 = na0; a1 = na1; b0 = nb0;
        if constexpr (NT == 2) b1 = nb1;
    }
    // epilogue: last K tile
    c00 = wmma16(a0, b0, c00);
    c10 = wmma16(a1, b0, c10);
    if constexpr (NT == 2) {
        c01 = wmma16(a0, b1, c01);
        c11 = wmma16(a1, b1, c11);
    }

    // C/D layout: vgpr r -> M = 8*laneHi + r, N = laneLo.
    float* o00 = Hagg + ((size_t)b * NPTS + m0 + laneHi * 8) * DPAD + n0 + laneLo;
    float* o10 = o00 + (size_t)16 * DPAD;
#pragma unroll
    for (int r = 0; r < 8; ++r) o00[(size_t)r * DPAD] = c00[r];
#pragma unroll
    for (int r = 0; r < 8; ++r) o10[(size_t)r * DPAD] = c10[r];
    if constexpr (NT == 2) {
#pragma unroll
        for (int r = 0; r < 8; ++r) o00[(size_t)r * DPAD + 16] = c01[r];
#pragma unroll
        for (int r = 0; r < 8; ++r) o10[(size_t)r * DPAD + 16] = c11[r];
    }
}

// -------------------- y = (Hagg/S) @ W + b ; LayerNorm ; swish ; write HT^T
// blockDim.x == dout (64/128/256), one block per (i,b) row.
__global__ void dense_ln_swish_kernel(const float* __restrict__ Hagg,
                                      const float* __restrict__ S,
                                      const float* __restrict__ W,
                                      const float* __restrict__ bias,
                                      const float* __restrict__ g,
                                      const float* __restrict__ be,
                                      f16* __restrict__ HTout,
                                      int din, int ld, int dout) {
    int i = blockIdx.x, b = blockIdx.y, j = threadIdx.x;
    __shared__ float sh[128];
    __shared__ float red[256];

    const float* hrow = Hagg + ((size_t)b * NPTS + i) * ld;
    for (int k = j; k < din; k += blockDim.x) sh[k] = hrow[k];
    __syncthreads();

    float invS = 1.0f / S[b * NPTS + i];
    float y = 0.f;
    for (int k = 0; k < din; ++k) y += sh[k] * W[(size_t)k * dout + j];
    y = y * invS + bias[j];

    // LayerNorm over dout
    red[j] = y; __syncthreads();
    for (int s = blockDim.x >> 1; s > 0; s >>= 1) {
        if (j < s) red[j] += red[j+s];
        __syncthreads();
    }
    float mu = red[0] / (float)dout;
    __syncthreads();
    float d = y - mu;
    red[j] = d * d; __syncthreads();
    for (int s = blockDim.x >> 1; s > 0; s >>= 1) {
        if (j < s) red[j] += red[j+s];
        __syncthreads();
    }
    float var = red[0] / (float)dout;
    float yn = d * rsqrtf(var + 1e-6f) * g[j] + be[j];
    float sw = yn / (1.0f + __expf(-yn));                  // swish
    HTout[((size_t)b * dout + j) * NPTS + i] = (f16)sw;    // store transposed
}

// ------------------------------- masked mean pool + final Dense(latent)
__global__ void pool_dense_kernel(const f16* __restrict__ HT,      // (B,256,N)
                                  const float* __restrict__ mask,
                                  const float* __restrict__ cent,
                                  const float* __restrict__ Wz,
                                  const float* __restrict__ bz,
                                  float* __restrict__ out) {
    int b = blockIdx.x, t = threadIdx.x;   // t = feature index (256)
    __shared__ float gf[256];
    const f16*   h = HT + ((size_t)b * 256 + t) * NPTS;
    const float* m = mask + (size_t)b * NPTS;
    float acc = 0.f;
    for (int n = 0; n < NPTS; ++n) acc += (float)h[n] * m[n];
    gf[t] = acc / fmaxf(cent[b*4+3], 1.0f);
    __syncthreads();
    if (t < LATENT) {
        float z = bz[t];
        for (int d = 0; d < 256; ++d) z += gf[d] * Wz[(size_t)d * LATENT + t];
        out[b * LATENT + t] = z;
    }
}

// ---------------------------------------------------------------- launcher
extern "C" void kernel_launch(void* const* d_in, const int* in_sizes, int n_in,
                              void* d_out, int out_size, void* d_ws, size_t ws_size,
                              hipStream_t stream) {
    (void)in_sizes; (void)n_in; (void)out_size; (void)ws_size;
    const float* x    = (const float*)d_in[0];
    const float* mask = (const float*)d_in[1];
    const float* W0 = (const float*)d_in[2];
    const float* b0 = (const float*)d_in[3];
    const float* g0 = (const float*)d_in[4];
    const float* e0 = (const float*)d_in[5];
    const float* W1 = (const float*)d_in[6];
    const float* b1 = (const float*)d_in[7];
    const float* g1 = (const float*)d_in[8];
    const float* e1 = (const float*)d_in[9];
    const float* W2 = (const float*)d_in[10];
    const float* b2 = (const float*)d_in[11];
    const float* g2 = (const float*)d_in[12];
    const float* e2 = (const float*)d_in[13];
    const float* Wz = (const float*)d_in[14];
    const float* bz = (const float*)d_in[15];
    float* out = (float*)d_out;

    // workspace carve-up (all 256B aligned); total ~90MB
    char* ws = (char*)d_ws;
    size_t off = 0;
    float* cent = (float*)(ws + off); off += 256;
    float* xc4  = (float*)(ws + off); off += (size_t)BATCH * NPTS * 4 * sizeof(float);
    float* S    = (float*)(ws + off); off += (size_t)BATCH * NPTS * sizeof(float);
    f16*   P    = (f16*)(ws + off);   off += (size_t)BATCH * NPTS * NPTS * sizeof(f16);
    f16*   HT0  = (f16*)(ws + off);   off += (size_t)BATCH * 16  * NPTS * sizeof(f16);
    f16*   HT1  = (f16*)(ws + off);   off += (size_t)BATCH * 64  * NPTS * sizeof(f16);
    f16*   HT2  = (f16*)(ws + off);   off += (size_t)BATCH * 128 * NPTS * sizeof(f16);
    f16*   HT3  = (f16*)(ws + off);   off += (size_t)BATCH * 256 * NPTS * sizeof(f16);
    float* Hagg = (float*)(ws + off); off += (size_t)BATCH * NPTS * 128 * sizeof(float);

    centroid_kernel<<<BATCH, 256, 0, stream>>>(x, mask, cent);
    center_kernel<<<(BATCH * NPTS) / 256, 256, 0, stream>>>(x, cent, xc4, HT0);
    distexp_kernel<<<dim3(NPTS, BATCH), 256, 0, stream>>>(xc4, P, S);

    // layer 0: din 3 (padded 16) -> 64
    agg_wmma_kernel<16, 1><<<dim3(NPTS / 128, 1, BATCH), 128, 0, stream>>>(P, HT0, Hagg);
    dense_ln_swish_kernel<<<dim3(NPTS, BATCH), 64, 0, stream>>>(
        Hagg, S, W0, b0, g0, e0, HT1, 3, 16, 64);

    // layer 1: 64 -> 128
    agg_wmma_kernel<64, 2><<<dim3(NPTS / 128, 2, BATCH), 128, 0, stream>>>(P, HT1, Hagg);
    dense_ln_swish_kernel<<<dim3(NPTS, BATCH), 128, 0, stream>>>(
        Hagg, S, W1, b1, g1, e1, HT2, 64, 64, 128);

    // layer 2: 128 -> 256
    agg_wmma_kernel<128, 2><<<dim3(NPTS / 128, 4, BATCH), 128, 0, stream>>>(P, HT2, Hagg);
    dense_ln_swish_kernel<<<dim3(NPTS, BATCH), 256, 0, stream>>>(
        Hagg, S, W2, b2, g2, e2, HT3, 128, 128, 256);

    pool_dense_kernel<<<BATCH, 256, 0, stream>>>(HT3, mask, cent, Wz, bz, out);
}
